// CheatDetector_45140106281723
// MI455X (gfx1250) — matrix operand order repaired and checked by reading, not compiled, not run on hardware
//
#include <hip/hip_runtime.h>
#include <hip/hip_bf16.h>

typedef __attribute__((ext_vector_type(16))) _Float16 v16h;
typedef __attribute__((ext_vector_type(8)))  float    v8f;

#define T_STEPS 512
#define DIN     4
#define H1      64
#define G1      256
#define H2      32
#define G2      128
#define BT      32      // batch rows per block
#define GSTRIDE 260     // padded gate-row stride (floats): 260 mod 64 banks = 4
#define H1S     72      // padded sH1 row stride (halves), 144 B (16B aligned)
#define H2S     40      // padded sH2 row stride (halves), 80 B (16B aligned)

__device__ __forceinline__ float sigmoidf_fast(float x) {
    return 1.0f / (1.0f + __expf(-x));
}

// A fragment (16x32, f16) per CDNA5 ISA 7.12.2:
// lanes 0-15 : M=lane, halves 0..7 -> K=kBase+0..7,  halves 8..15 -> K=kBase+16..23
// lanes 16-31: M=lane-16, halves 0..7 -> K=kBase+8..15, halves 8..15 -> K=kBase+24..31
__device__ __forceinline__ v16h load_a_frag(const _Float16* __restrict__ s, int stride,
                                            int mBase, int kBase, int lane) {
    const _Float16* p = s + (mBase + (lane & 15)) * stride + kBase + ((lane & 16) ? 8 : 0);
    v16h a;
#pragma unroll
    for (int h = 0; h < 8; ++h) { a[h] = p[h]; a[h + 8] = p[16 + h]; }
    return a;
}

// B fragment (32x16, f16) for gates = h @ W^T  (B[k][n] = W[nBase+n][kBase+k]):
// lanes 0-15: N=lane, halves 0..7 -> K=kBase+0..7, 8..15 -> K=kBase+16..23 ; lanes 16-31: N=lane-16, halves 0..7 -> K=kBase+8..15, 8..15 -> K=kBase+24..31
__device__ __forceinline__ v16h load_b_frag(const float* __restrict__ W, int ldk,
                                            int nBase, int kBase, int lane) {
    const float* p = W + (size_t)(nBase + (lane & 15)) * ldk + kBase + ((lane & 16) ? 8 : 0);
    v16h b;
#pragma unroll
    for (int h = 0; h < 8; ++h) { b[h] = (_Float16)p[h]; b[h + 8] = (_Float16)p[16 + h]; }
    return b;
}

__global__ __launch_bounds__(256, 1)
void lstm2_fused_wmma(const float* __restrict__ x,
                      const float* __restrict__ w_ih1, const float* __restrict__ w_hh1,
                      const float* __restrict__ b_ih1, const float* __restrict__ b_hh1,
                      const float* __restrict__ w_ih2, const float* __restrict__ w_hh2,
                      const float* __restrict__ b_ih2, const float* __restrict__ b_hh2,
                      const float* __restrict__ w_fc1, const float* __restrict__ b_fc1,
                      const float* __restrict__ w_fc2, const float* __restrict__ b_fc2,
                      float* __restrict__ out)
{
    __shared__ _Float16 sH1[BT * H1S];       // layer-1 hidden state (f16 for WMMA A)
    __shared__ _Float16 sH2[BT * H2S];       // layer-2 hidden state
    __shared__ float    sC1[BT * H1];        // cell states (fp32)
    __shared__ float    sC2[BT * H2];
    __shared__ float    sG [BT * GSTRIDE];   // gate scratch, reused by both layers
    __shared__ float    sX [BT * DIN];       // x[:, t, :] tile
    __shared__ float    sWih1[G1 * DIN];     // tiny K=4 input projection (VALU path)
    __shared__ float    sB1[G1];             // b_ih1 + b_hh1
    __shared__ float    sB2[G2];             // b_ih2 + b_hh2
    __shared__ float    sWfc1[16 * H2];
    __shared__ float    sBfc1[16];
    __shared__ float    sWfc2[16];

    const int tid  = threadIdx.x;
    const int lane = tid & 31;
    const int wv   = tid >> 5;               // wave id 0..7
    const int row0 = blockIdx.x * BT;        // first batch row of this block

    // ---------------- one-time LDS staging ----------------
    for (int i = tid; i < G1 * DIN; i += 256) sWih1[i] = w_ih1[i];
    for (int i = tid; i < G1;       i += 256) sB1[i]   = b_ih1[i] + b_hh1[i];
    for (int i = tid; i < G2;       i += 256) sB2[i]   = b_ih2[i] + b_hh2[i];
    for (int i = tid; i < 16 * H2;  i += 256) sWfc1[i] = w_fc1[i];
    if (tid < 16) { sBfc1[tid] = b_fc1[tid]; sWfc2[tid] = w_fc2[tid]; }
    for (int i = tid; i < BT * H1S; i += 256) sH1[i] = (_Float16)0.f;
    for (int i = tid; i < BT * H2S; i += 256) sH2[i] = (_Float16)0.f;
    for (int i = tid; i < BT * H1;  i += 256) sC1[i] = 0.f;
    for (int i = tid; i < BT * H2;  i += 256) sC2[i] = 0.f;

    // -------- persistent weight B-fragments (registers, reused 512 steps) --------
    // layer-1: wave wv owns gate cols [32*wv, 32*wv+32): 2 n-tiles x (K=64 -> 2 chunks)
    v16h b1f[2][2];
#pragma unroll
    for (int nt = 0; nt < 2; ++nt)
#pragma unroll
        for (int kc = 0; kc < 2; ++kc)
            b1f[nt][kc] = load_b_frag(w_hh1, H1, 32 * wv + 16 * nt, 32 * kc, lane);
    // layer-2: wave wv owns gate cols [16*wv, 16*wv+16): K = 64 (h1) + 32 (h2)
    v16h b2f[3];
    b2f[0] = load_b_frag(w_ih2, H1, 16 * wv, 0,  lane);
    b2f[1] = load_b_frag(w_ih2, H1, 16 * wv, 32, lane);
    b2f[2] = load_b_frag(w_hh2, H2, 16 * wv, 0,  lane);

    __syncthreads();

    // per-lane column constants (this lane always writes the same gate column n)
    float wx1[2][DIN], rb1[2];
#pragma unroll
    for (int nt = 0; nt < 2; ++nt) {
        int n = 32 * wv + 16 * nt + (lane & 15);
        rb1[nt] = sB1[n];
#pragma unroll
        for (int d = 0; d < DIN; ++d) wx1[nt][d] = sWih1[n * DIN + d];
    }
    const float rb2 = sB2[16 * wv + (lane & 15)];

    // ---------------- the sequential scan ----------------
#pragma unroll 1
    for (int t = 0; t < T_STEPS; ++t) {
        // stage x[:, t, :] tile (128 floats)
        if (tid < BT * DIN) {
            int r = tid >> 2, d = tid & 3;
            sX[tid] = x[((size_t)(row0 + r) * T_STEPS + t) * DIN + d];
        }
        __syncthreads();  // x ready; prev-step h2/sG consumers done

        // ---- layer-1 gates: sG[:, 0:256] = h1 @ w_hh1^T + x @ w_ih1^T + b ----
#pragma unroll
        for (int mt = 0; mt < BT / 16; ++mt) {
            v16h a0 = load_a_frag(sH1, H1S, mt * 16, 0,  lane);
            v16h a1 = load_a_frag(sH1, H1S, mt * 16, 32, lane);
#pragma unroll
            for (int nt = 0; nt < 2; ++nt) {
                v8f acc = {};
                acc = __builtin_amdgcn_wmma_f32_16x16x32_f16(false, a0, false, b1f[nt][0],
                                                             (short)0, acc, false, false);
                acc = __builtin_amdgcn_wmma_f32_16x16x32_f16(false, a1, false, b1f[nt][1],
                                                             (short)0, acc, false, false);
                const int n  = 32 * wv + 16 * nt + (lane & 15);
                const int mb = mt * 16 + ((lane & 16) ? 8 : 0);
#pragma unroll
                for (int r2 = 0; r2 < 8; ++r2) {           // C/D layout: VGPR r -> M=mb+r
                    int m = mb + r2;
                    float g = acc[r2] + rb1[nt];
#pragma unroll
                    for (int d = 0; d < DIN; ++d) g += sX[m * DIN + d] * wx1[nt][d];
                    sG[m * GSTRIDE + n] = g;
                }
            }
        }
        __syncthreads();

        // ---- layer-1 pointwise: c1,h1 update ----
        for (int p = tid; p < BT * H1; p += 256) {
            int m = p >> 6, j = p & 63;
            const float* gr = sG + m * GSTRIDE;
            float ig = sigmoidf_fast(gr[j]);
            float fg = sigmoidf_fast(gr[64 + j]);
            float gg = tanhf(gr[128 + j]);
            float og = sigmoidf_fast(gr[192 + j]);
            float c  = fg * sC1[p] + ig * gg;
            sC1[p] = c;
            sH1[m * H1S + j] = (_Float16)(og * tanhf(c));
        }
        __syncthreads();

        // ---- layer-2 gates: sG[:, 0:128] = h1 @ w_ih2^T + h2 @ w_hh2^T + b ----
#pragma unroll
        for (int mt = 0; mt < BT / 16; ++mt) {
            v16h a0 = load_a_frag(sH1, H1S, mt * 16, 0,  lane);
            v16h a1 = load_a_frag(sH1, H1S, mt * 16, 32, lane);
            v16h a2 = load_a_frag(sH2, H2S, mt * 16, 0,  lane);
            v8f acc = {};
            acc = __builtin_amdgcn_wmma_f32_16x16x32_f16(false, a0, false, b2f[0],
                                                         (short)0, acc, false, false);
            acc = __builtin_amdgcn_wmma_f32_16x16x32_f16(false, a1, false, b2f[1],
                                                         (short)0, acc, false, false);
            acc = __builtin_amdgcn_wmma_f32_16x16x32_f16(false, a2, false, b2f[2],
                                                         (short)0, acc, false, false);
            const int n  = 16 * wv + (lane & 15);
            const int mb = mt * 16 + ((lane & 16) ? 8 : 0);
#pragma unroll
            for (int r2 = 0; r2 < 8; ++r2)
                sG[(mb + r2) * GSTRIDE + n] = acc[r2] + rb2;
        }
        __syncthreads();

        // ---- layer-2 pointwise: c2,h2 update ----
        for (int p = tid; p < BT * H2; p += 256) {
            int m = p >> 5, j = p & 31;
            const float* gr = sG + m * GSTRIDE;
            float ig = sigmoidf_fast(gr[j]);
            float fg = sigmoidf_fast(gr[32 + j]);
            float gg = tanhf(gr[64 + j]);
            float og = sigmoidf_fast(gr[96 + j]);
            float c  = fg * sC2[p] + ig * gg;
            sC2[p] = c;
            sH2[m * H2S + j] = (_Float16)(og * tanhf(c));
        }
        __syncthreads();
    }

    // ---------------- FC head on final h2 ----------------
    if (tid < BT) {
        float accum = b_fc2[0];
#pragma unroll
        for (int k = 0; k < 16; ++k) {
            float z = sBfc1[k];
#pragma unroll
            for (int j = 0; j < H2; ++j)
                z += (float)sH2[tid * H2S + j] * sWfc1[k * H2 + j];
            accum += fmaxf(z, 0.f) * sWfc2[k];
        }
        out[row0 + tid] = accum;
    }
}

extern "C" void kernel_launch(void* const* d_in, const int* in_sizes, int n_in,
                              void* d_out, int out_size, void* d_ws, size_t ws_size,
                              hipStream_t stream) {
    (void)n_in; (void)d_ws; (void)ws_size; (void)out_size;
    const float* x     = (const float*)d_in[0];
    const float* w_ih1 = (const float*)d_in[1];
    const float* w_hh1 = (const float*)d_in[2];
    const float* b_ih1 = (const float*)d_in[3];
    const float* b_hh1 = (const float*)d_in[4];
    const float* w_ih2 = (const float*)d_in[5];
    const float* w_hh2 = (const float*)d_in[6];
    const float* b_ih2 = (const float*)d_in[7];
    const float* b_hh2 = (const float*)d_in[8];
    const float* w_fc1 = (const float*)d_in[9];
    const float* b_fc1 = (const float*)d_in[10];
    const float* w_fc2 = (const float*)d_in[11];
    const float* b_fc2 = (const float*)d_in[12];
    float* out = (float*)d_out;

    const int batch = in_sizes[0] / (T_STEPS * DIN);   // 4096
    const int grid  = batch / BT;                      // 128 blocks

    lstm2_fused_wmma<<<dim3(grid), dim3(256), 0, stream>>>(
        x, w_ih1, w_hh1, b_ih1, b_hh1,
        w_ih2, w_hh2, b_ih2, b_hh2,
        w_fc1, b_fc1, w_fc2, b_fc2, out);
}
